// Node2Pair_bias_42528766165481
// MI455X (gfx1250) — matrix operand order repaired and checked by reading, not compiled, not run on hardware
//
#include <hip/hip_runtime.h>

// Sizes from the reference
#define D_NODE   256
#define D_HIDDEN 32
#define D_PAIR   128
#define D_HEAD   16
#define NB       2
#define NL       512
#define LN_EPS   1e-5f

typedef __attribute__((ext_vector_type(16))) __bf16 v16bf;
typedef __attribute__((ext_vector_type(8)))  __bf16 v8bf;
typedef __attribute__((ext_vector_type(8)))  float  v8f;

// -------------------------------------------------------------------------
// Phase 1: per (b,l) row: LayerNorm + mask, left/right projections.
// Emits:
//   leftbf[(b*512+l)*32 + c]            = bf16(left[c])
//   Mbf  [(b*512+l)*4096 + p*32 + c]    = bf16(right_scaled[c] * W_out[c>>1][p])
// (M stored TRANSPOSED [p][c] so phase-2 B-fragments are contiguous 32B loads)
// -------------------------------------------------------------------------
__global__ void __launch_bounds__(256) n2p_phase1(
    const float* __restrict__ node, const unsigned char* __restrict__ mask,
    const float* __restrict__ gamma, const float* __restrict__ beta,
    const float* __restrict__ Wl, const float* __restrict__ bl,
    const float* __restrict__ Wr, const float* __restrict__ br,
    const float* __restrict__ Wout,
    __bf16* __restrict__ leftbf, __bf16* __restrict__ Mbf)
{
    __shared__ float sx[256];
    __shared__ float r1[256];
    __shared__ float r2[256];
    __shared__ float sr[32];

    const int t   = threadIdx.x;
    const int row = blockIdx.x;                 // b*512 + l

    float x = node[(size_t)row * 256 + t];
    r1[t] = x; r2[t] = x * x;
    __syncthreads();
    for (int s = 128; s > 0; s >>= 1) {
        if (t < s) { r1[t] += r1[t + s]; r2[t] += r2[t + s]; }
        __syncthreads();
    }
    const float mu  = r1[0] * (1.0f / 256.0f);
    const float var = r2[0] * (1.0f / 256.0f) - mu * mu;
    const float inv = rsqrtf(var + LN_EPS);
    const float mk  = mask[row] ? 1.0f : 0.0f;

    sx[t] = ((x - mu) * inv * gamma[t] + beta[t]) * mk;
    __syncthreads();

    if (t < 64) {
        const int c = t & 31;
        const float* __restrict__ W = (t < 32) ? Wl : Wr;
        float acc = (t < 32) ? bl[c] : br[c];
        #pragma unroll 8
        for (int d = 0; d < 256; ++d) acc += sx[d] * W[d * 32 + c];
        if (t < 32) {
            leftbf[(size_t)row * 32 + c] = (__bf16)acc;
        } else {
            sr[c] = acc * 0.17677669529663688f;   // 1/sqrt(D_HIDDEN)
        }
    }
    __syncthreads();

    // M_T[p][c] = sr[c] * Wout[c>>1][p]; thread t covers p=t>>1, c in 16-chunk
    __bf16* __restrict__ M = Mbf + (size_t)row * 4096;
    const int p  = t >> 1;
    const int c0 = (t & 1) * 16;
    #pragma unroll
    for (int cc = 0; cc < 16; ++cc) {
        const int c = c0 + cc;
        M[p * 32 + c] = (__bf16)(sr[c] * Wout[(c >> 1) * 128 + p]);
    }
}

// -------------------------------------------------------------------------
// Phase 2: one block per (b,j). out[b,i,j,p] = left[b,i,:] @ M_j[:,p] + b_out[p]
// 8 waves; wave owns 4 i-tiles x 8 p-tiles; K=32 in ONE v_wmma_f32_16x16x32_bf16.
// -------------------------------------------------------------------------
__global__ void __launch_bounds__(256) n2p_phase2(
    const __bf16* __restrict__ leftbf, const __bf16* __restrict__ Mbf,
    const float* __restrict__ bout, float* __restrict__ out)
{
    const int t    = threadIdx.x;
    const int lane = t & 31;
    const int wave = t >> 5;
    const int bj   = blockIdx.x;                // b*512 + j
    const int b    = bj >> 9;
    const int j    = bj & 511;
    const int lh   = lane >> 4;                 // lane half
    const int n    = lane & 15;                 // row-in-half / N column

    // B fragments (reused for all i-tiles): M_T row = pt*16+n, cols lh*16..+15
    const __bf16* __restrict__ Mrow = Mbf + (size_t)bj * 4096;
    v16bf bfrag[8];
    #pragma unroll
    for (int pt = 0; pt < 8; ++pt) {
        const __bf16* src = Mrow + (pt * 16 + n) * 32 + lh * 16;
        v8bf lo = *(const v8bf*)(src);
        v8bf hi = *(const v8bf*)(src + 8);
        bfrag[pt] = __builtin_shufflevector(lo, hi,
            0,1,2,3,4,5,6,7,8,9,10,11,12,13,14,15);
    }

    #pragma unroll
    for (int q = 0; q < 4; ++q) {
        const int itile = wave * 4 + q;
        const int i     = itile * 16 + n;       // A row for this lane
        // A layout: lanes 0-15 K={0..15}; lanes 16-31 K={16..31}
        const __bf16* arow = leftbf + ((size_t)(b * 512 + i)) * 32 + lh * 16;
        v8bf alo = *(const v8bf*)(arow);        // K = lh*16 .. lh*16+7
        v8bf ahi = *(const v8bf*)(arow + 8);    // K = lh*16+8 .. lh*16+15
        v16bf a = __builtin_shufflevector(alo, ahi,
            0,1,2,3,4,5,6,7,8,9,10,11,12,13,14,15);

        #pragma unroll
        for (int pt = 0; pt < 8; ++pt) {
            // Accumulator pre-loaded with bias: lane's p is fixed across VGPRs
            const float bias = bout[pt * 16 + n];
            v8f c;
            #pragma unroll
            for (int v = 0; v < 8; ++v) c[v] = bias;

            v8f d = __builtin_amdgcn_wmma_f32_16x16x32_bf16(
                false, a, false, bfrag[pt], (short)0, c, false, false);

            // D layout: row M = v + 8*lh, col N = n
            const size_t i0   = (size_t)(b * 512 + itile * 16 + lh * 8);
            const size_t base = ((i0 * 512 + (size_t)j) * 128) + pt * 16 + n;
            #pragma unroll
            for (int v = 0; v < 8; ++v)
                out[base + (size_t)v * (512 * 128)] = d[v];
        }
    }
}

extern "C" void kernel_launch(void* const* d_in, const int* in_sizes, int n_in,
                              void* d_out, int out_size, void* d_ws, size_t ws_size,
                              hipStream_t stream)
{
    const float*         node  = (const float*)d_in[0];
    const unsigned char* mask  = (const unsigned char*)d_in[1];
    const float*         gamma = (const float*)d_in[2];
    const float*         beta  = (const float*)d_in[3];
    const float*         Wl    = (const float*)d_in[4];
    const float*         bl    = (const float*)d_in[5];
    const float*         Wr    = (const float*)d_in[6];
    const float*         br    = (const float*)d_in[7];
    const float*         Wout  = (const float*)d_in[8];
    const float*         bout  = (const float*)d_in[9];
    float*               out   = (float*)d_out;

    // Workspace layout: M_T bf16 [1024][128][32] = 8 MB, then left bf16 64 KB
    __bf16* Mbf    = (__bf16*)d_ws;
    __bf16* leftbf = (__bf16*)((char*)d_ws + (size_t)NB * NL * 4096 * sizeof(__bf16));

    n2p_phase1<<<NB * NL, 256, 0, stream>>>(node, mask, gamma, beta,
                                            Wl, bl, Wr, br, Wout, leftbf, Mbf);
    n2p_phase2<<<NB * NL, 256, 0, stream>>>(leftbf, Mbf, bout, out);
}
